// ConformerBlock_31542239822419
// MI455X (gfx1250) — matrix operand order repaired and checked by reading, not compiled, not run on hardware
//
#include <hip/hip_runtime.h>
#include <hip/hip_bf16.h>

typedef __attribute__((ext_vector_type(16))) _Float16 v16h;
typedef __attribute__((ext_vector_type(8)))  float    v8f;
typedef __attribute__((ext_vector_type(8)))  _Float16 h8;   // 16B aligned
typedef __attribute__((ext_vector_type(4)))  _Float16 h4;

#define DIM   768
#define SEQ   1024
#define BATCH 8
#define HEADS 12
#define HDIM  64
#define ROWS  (BATCH*SEQ)   /* 8192 */

// ---------------------------------------------------------------- utilities

__device__ __forceinline__ float blockReduceSum(float v, float* sh) {
#pragma unroll
  for (int d = 16; d >= 1; d >>= 1) v += __shfl_xor(v, d, 32);
  int w = threadIdx.x >> 5;
  if ((threadIdx.x & 31) == 0) sh[w] = v;
  __syncthreads();
  float r = 0.f;
  if (threadIdx.x < 8) r = sh[threadIdx.x];
  if (threadIdx.x < 32) {
#pragma unroll
    for (int d = 4; d >= 1; d >>= 1) r += __shfl_xor(r, d, 32);
  }
  if (threadIdx.x == 0) sh[0] = r;
  __syncthreads();
  r = sh[0];
  __syncthreads();
  return r;
}

__device__ __forceinline__ v16h frag_from(const _Float16* p0, const _Float16* p1) {
  // 16 f16 fragment from two 16B-aligned contiguous halves
  h8 lo = *(const h8*)p0;
  h8 hi = *(const h8*)p1;
  v16h f;
#pragma unroll
  for (int e = 0; e < 8; ++e) { f[e] = lo[e]; f[8 + e] = hi[e]; }
  return f;
}

// ------------------------------------------------------------ weight cast

__global__ void convert_w_kernel(const float* __restrict__ src,
                                 _Float16* __restrict__ dst,
                                 int rows, int cols, int transpose) {
  size_t n = (size_t)rows * cols;
  for (size_t i = (size_t)blockIdx.x * blockDim.x + threadIdx.x; i < n;
       i += (size_t)gridDim.x * blockDim.x) {
    if (!transpose) {
      dst[i] = (_Float16)src[i];
    } else {
      size_t r = i / cols, c = i % cols;
      dst[c * (size_t)rows + r] = (_Float16)src[i];
    }
  }
}

// ------------------------------------------------------------ layer norm

__global__ __launch_bounds__(256)
void layernorm_kernel(const float* __restrict__ x, const float* __restrict__ g,
                      const float* __restrict__ bta,
                      _Float16* __restrict__ out16, float* __restrict__ out32) {
  __shared__ float sh[8];
  const int row = blockIdx.x;
  const int tid = threadIdx.x;
  const float* xr = x + (size_t)row * DIM;
  float s = 0.f, ss = 0.f;
  for (int i = tid; i < DIM; i += 256) { float v = xr[i]; s += v; ss += v * v; }
  s = blockReduceSum(s, sh);
  ss = blockReduceSum(ss, sh);
  float mean = s * (1.0f / DIM);
  float var  = ss * (1.0f / DIM) - mean * mean;
  float inv  = rsqrtf(var + 1e-5f);
  for (int i = tid; i < DIM; i += 256) {
    float v = (xr[i] - mean) * inv * g[i] + bta[i];
    if (out16) out16[(size_t)row * DIM + i] = (_Float16)v;
    else       out32[(size_t)row * DIM + i] = v;
  }
}

// ------------------------------------------------------------ WMMA GEMM
// C(M,N) = A(M,K) @ B(K,N); block tile 128x64, 4 waves, wave = 32(M) x 64(N).
// A staged row-major, B staged COLUMN-major in LDS so B fragments are
// contiguous (2 x ds_load_b128 per fragment).
// MODE: 0 bias->f16 | 1 bias+GELU->f16 | 2 bias->f32 | 3 res + scale*(C+bias)->f32

#define GA_STR 40   // [m][k] row stride (f16): 32 + 8 pad, 16B-aligned frags
#define GBT_STR 40  // [n][k] column stride

template <int MODE>
__global__ __launch_bounds__(128)
void gemm_wmma_kernel(const _Float16* __restrict__ A,
                      const _Float16* __restrict__ Bw,
                      const float* __restrict__ bias,
                      const float* res, float scale,
                      _Float16* out16, float* out32,
                      int Kdim, int Ncols) {
  __shared__ _Float16 As[128 * GA_STR];
  __shared__ _Float16 BsT[64 * GBT_STR];
  const int tid  = threadIdx.x;
  const int lane = tid & 31;
  const int wv   = tid >> 5;
  const int l16  = lane & 15;
  const int half = lane >> 4;
  const int mBase = blockIdx.y * 128;
  const int nBase = blockIdx.x * 64;

  v8f acc[2][4] = {};

  for (int k0 = 0; k0 < Kdim; k0 += 32) {
    __syncthreads();
#pragma unroll
    for (int q = 0; q < 4; ++q) {           // A tile: 128x32 f16, 512 h8 loads
      int qi = tid + q * 128;               // 0..511
      int r = qi >> 2, c8 = (qi & 3) << 3;
      *(h8*)&As[r * GA_STR + c8] =
          *(const h8*)(A + (size_t)(mBase + r) * Kdim + k0 + c8);
    }
#pragma unroll
    for (int q = 0; q < 2; ++q) {           // B tile: 32x64, transpose-scatter
      int qi = tid + q * 128;               // 0..255
      int r = qi >> 3, c8 = (qi & 7) << 3;
      h8 v = *(const h8*)(Bw + (size_t)(k0 + r) * Ncols + nBase + c8);
#pragma unroll
      for (int j = 0; j < 8; ++j) BsT[(c8 + j) * GBT_STR + r] = v[j];
    }
    __syncthreads();

    // two A fragments per wave (rows wv*32 .. wv*32+31)
    const _Float16* ar0 = &As[(wv * 32 + l16) * GA_STR + half * 8];
    const _Float16* ar1 = ar0 + 16 * GA_STR;
    v16h a0 = frag_from(ar0, ar0 + 16);
    v16h a1 = frag_from(ar1, ar1 + 16);
#pragma unroll
    for (int nt = 0; nt < 4; ++nt) {
      const _Float16* br = &BsT[(nt * 16 + l16) * GBT_STR + half * 8];
      v16h b = frag_from(br, br + 16);
      acc[0][nt] = __builtin_amdgcn_wmma_f32_16x16x32_f16(
          false, a0, false, b, (short)0, acc[0][nt], false, false);
      acc[1][nt] = __builtin_amdgcn_wmma_f32_16x16x32_f16(
          false, a1, false, b, (short)0, acc[1][nt], false, false);
    }
  }

#pragma unroll
  for (int mt = 0; mt < 2; ++mt) {
    const int rowOff = mBase + wv * 32 + mt * 16 + half * 8;  // M = r + 8*half
#pragma unroll
    for (int nt = 0; nt < 4; ++nt) {
      int col = nBase + nt * 16 + l16;
      float bv = bias[col];
#pragma unroll
      for (int r = 0; r < 8; ++r) {
        size_t idx = (size_t)(rowOff + r) * Ncols + col;
        float v = acc[mt][nt][r] + bv;
        if (MODE == 1) v = 0.5f * v * (1.0f + erff(v * 0.70710678f));
        if (MODE == 0 || MODE == 1) out16[idx] = (_Float16)v;
        else if (MODE == 2)         out32[idx] = v;
        else                        out32[idx] = res[idx] + scale * v;
      }
    }
  }
}

// ------------------------------------------------------- flash attention
// grid = (SEQ/128, B*H); 8 waves/block, each wave owns a 16-row Q tile.
// 64-key slabs: 4 S-tiles per online-softmax pass (halves shuffle-reduction
// and rescale overhead per key vs 32-key slabs).
// K staged [key][dim] (fragments contiguous), V staged [dim][key] transposed.

#define KS_STR 72   // key row stride (64 + 8)
#define VT_STR 72   // dim row stride (64 keys + 8)
#define PS_STR 72   // P row stride   (64 keys + 8)

__global__ __launch_bounds__(256)
void attn_flash_kernel(const _Float16* __restrict__ qkv,
                       const float* __restrict__ rel_table,
                       _Float16* __restrict__ outp) {
  __shared__ _Float16 Ks[64 * KS_STR];
  __shared__ _Float16 VsT[64 * VT_STR];
  __shared__ _Float16 Ps[8 * 16 * PS_STR];
  const int tid  = threadIdx.x;
  const int lane = tid & 31;
  const int wv   = tid >> 5;
  const int l16  = lane & 15;
  const int half = lane >> 4;
  const int b = blockIdx.y / HEADS;
  const int h = blockIdx.y % HEADS;
  const int qBase = blockIdx.x * 128 + wv * 16;
  const _Float16* qkvB = qkv + (size_t)b * SEQ * (3 * DIM);

  // Q A-fragments for hd K-slices [0,32) and [32,64)
  const _Float16* qr = qkvB + (size_t)(qBase + l16) * (3 * DIM) + h * HDIM + half * 8;
  v16h aq0 = frag_from(qr, qr + 16);
  v16h aq1 = frag_from(qr + 32, qr + 48);

  v8f o[4] = {};
  float Mr[8], Lr[8];
#pragma unroll
  for (int r = 0; r < 8; ++r) { Mr[r] = -3.0e38f; Lr[r] = 0.f; }
  _Float16* Pw = &Ps[wv * 16 * PS_STR];

  for (int kt = 0; kt < SEQ / 64; ++kt) {
    const int keyBase = kt * 64;
    __syncthreads();
#pragma unroll
    for (int q = 0; q < 2; ++q) {           // stage K,V 64x64 tiles (512 h8 each)
      int qi = tid + q * 256;               // 0..511
      int r = qi >> 3, c8 = (qi & 7) << 3;
      const _Float16* src =
          qkvB + (size_t)(keyBase + r) * (3 * DIM) + h * HDIM + c8;
      *(h8*)&Ks[r * KS_STR + c8] = *(const h8*)(src + DIM);
      h8 vq = *(const h8*)(src + 2 * DIM);
#pragma unroll
      for (int j = 0; j < 8; ++j) VsT[(c8 + j) * VT_STR + r] = vq[j];
    }
    __syncthreads();

    float sreg[4][8];
#pragma unroll
    for (int hf = 0; hf < 4; ++hf) {        // S tile (16 keys per quarter)
      const _Float16* kr = &Ks[(hf * 16 + l16) * KS_STR + half * 8];
      v16h bk0 = frag_from(kr, kr + 16);         // hd 0..31
      v16h bk1 = frag_from(kr + 32, kr + 48);    // hd 32..63
      v8f c = {};
      c = __builtin_amdgcn_wmma_f32_16x16x32_f16(false, aq0, false, bk0,
                                                 (short)0, c, false, false);
      c = __builtin_amdgcn_wmma_f32_16x16x32_f16(false, aq1, false, bk1,
                                                 (short)0, c, false, false);
      const int key = keyBase + hf * 16 + l16;
#pragma unroll
      for (int r = 0; r < 8; ++r) {
        int qrow = qBase + half * 8 + r;
        sreg[hf][r] = c[r] * 0.125f +
                      rel_table[(size_t)((key - qrow + SEQ - 1) * HEADS + h)];
      }
    }
    // online softmax over this 64-key slab
#pragma unroll
    for (int r = 0; r < 8; ++r) {
      float m = fmaxf(fmaxf(sreg[0][r], sreg[1][r]),
                      fmaxf(sreg[2][r], sreg[3][r]));
#pragma unroll
      for (int d = 8; d >= 1; d >>= 1) m = fmaxf(m, __shfl_xor(m, d, 32));
      float newM  = fmaxf(Mr[r], m);
      float alpha = __expf(Mr[r] - newM);
      Mr[r] = newM;
      float p0 = __expf(sreg[0][r] - newM);
      float p1 = __expf(sreg[1][r] - newM);
      float p2 = __expf(sreg[2][r] - newM);
      float p3 = __expf(sreg[3][r] - newM);
      float s = (p0 + p1) + (p2 + p3);
#pragma unroll
      for (int d = 8; d >= 1; d >>= 1) s += __shfl_xor(s, d, 32);
      Lr[r] = Lr[r] * alpha + s;
#pragma unroll
      for (int nt = 0; nt < 4; ++nt) o[nt][r] *= alpha;
      _Float16* prow = &Pw[(half * 8 + r) * PS_STR + l16];
      prow[0]  = (_Float16)p0;
      prow[16] = (_Float16)p1;
      prow[32] = (_Float16)p2;
      prow[48] = (_Float16)p3;
    }
    __syncthreads();
    // reload P in A-fragment layout, accumulate O += P @ V (K = 64 keys)
    const _Float16* pr = &Pw[l16 * PS_STR + half * 8];
    v16h ap0 = frag_from(pr, pr + 16);         // keys 0..31
    v16h ap1 = frag_from(pr + 32, pr + 48);    // keys 32..63
#pragma unroll
    for (int nt = 0; nt < 4; ++nt) {
      const _Float16* vr = &VsT[(nt * 16 + l16) * VT_STR + half * 8];
      v16h bv0 = frag_from(vr, vr + 16);       // keys 0..31
      v16h bv1 = frag_from(vr + 32, vr + 48);  // keys 32..63
      o[nt] = __builtin_amdgcn_wmma_f32_16x16x32_f16(false, ap0, false, bv0,
                                                     (short)0, o[nt], false, false);
      o[nt] = __builtin_amdgcn_wmma_f32_16x16x32_f16(false, ap1, false, bv1,
                                                     (short)0, o[nt], false, false);
    }
  }

#pragma unroll
  for (int nt = 0; nt < 4; ++nt) {
    int col = h * HDIM + nt * 16 + l16;
#pragma unroll
    for (int r = 0; r < 8; ++r) {
      int row = qBase + half * 8 + r;
      outp[(size_t)(b * SEQ + row) * DIM + col] = (_Float16)(o[nt][r] / Lr[r]);
    }
  }
}

// ------------------------------------------------------- conv module bits

__global__ void glu_kernel(const float* __restrict__ mid, float* __restrict__ g) {
  size_t total = (size_t)ROWS * DIM;
  for (size_t i = (size_t)blockIdx.x * blockDim.x + threadIdx.x; i < total;
       i += (size_t)gridDim.x * blockDim.x) {
    size_t row = i / DIM;
    int d = (int)(i % DIM);
    float u  = mid[row * (2 * DIM) + d];
    float gt = mid[row * (2 * DIM) + DIM + d];
    g[i] = u * (1.0f / (1.0f + __expf(-gt)));
  }
}

__global__ void dwconv_kernel(const float* __restrict__ g,
                              const float* __restrict__ w,
                              const float* __restrict__ bias,
                              float* __restrict__ out) {
  size_t total = (size_t)ROWS * DIM;
  for (size_t i = (size_t)blockIdx.x * blockDim.x + threadIdx.x; i < total;
       i += (size_t)gridDim.x * blockDim.x) {
    int d = (int)(i % DIM);
    size_t bn = i / DIM;
    int n = (int)(bn % SEQ);
    size_t bbase = (bn - (size_t)n) * DIM;
    float acc = bias[d];
#pragma unroll
    for (int t = 0; t < 9; ++t) {
      int nn = n + t - 4;
      if (nn >= 0 && nn < SEQ)
        acc += g[bbase + (size_t)nn * DIM + d] * w[d * 9 + t];
    }
    out[i] = acc;
  }
}

__global__ __launch_bounds__(256)
void bnstats_kernel(const float* __restrict__ y, float* __restrict__ mean,
                    float* __restrict__ inv) {
  __shared__ float sh[8];
  int d = blockIdx.x;
  float s = 0.f, ss = 0.f;
  for (int i = threadIdx.x; i < ROWS; i += 256) {
    float v = y[(size_t)i * DIM + d];
    s += v; ss += v * v;
  }
  s = blockReduceSum(s, sh);
  ss = blockReduceSum(ss, sh);
  if (threadIdx.x == 0) {
    float m = s * (1.0f / ROWS);
    mean[d] = m;
    inv[d] = rsqrtf(ss * (1.0f / ROWS) - m * m + 1e-5f);
  }
}

__global__ void bnsilu_kernel(const float* __restrict__ y,
                              const float* __restrict__ mean,
                              const float* __restrict__ inv,
                              const float* __restrict__ gg,
                              const float* __restrict__ bb,
                              _Float16* __restrict__ out16) {
  size_t total = (size_t)ROWS * DIM;
  for (size_t i = (size_t)blockIdx.x * blockDim.x + threadIdx.x; i < total;
       i += (size_t)gridDim.x * blockDim.x) {
    int d = (int)(i % DIM);
    float v = (y[i] - mean[d]) * inv[d] * gg[d] + bb[d];
    out16[i] = (_Float16)(v * (1.0f / (1.0f + __expf(-v))));
  }
}

// ---------------------------------------------------------------- driver

extern "C" void kernel_launch(void* const* d_in, const int* in_sizes, int n_in,
                              void* d_out, int out_size, void* d_ws, size_t ws_size,
                              hipStream_t stream) {
  (void)in_sizes; (void)n_in; (void)out_size; (void)ws_size;
  const float* x       = (const float*)d_in[0];
  const float* ff1_g   = (const float*)d_in[1];
  const float* ff1_b   = (const float*)d_in[2];
  const float* ff1_w1  = (const float*)d_in[3];
  const float* ff1_b1  = (const float*)d_in[4];
  const float* ff1_w2  = (const float*)d_in[5];
  const float* ff1_b2  = (const float*)d_in[6];
  const float* at_g    = (const float*)d_in[7];
  const float* at_b    = (const float*)d_in[8];
  const float* qkv_w   = (const float*)d_in[9];
  const float* qkv_b   = (const float*)d_in[10];
  const float* proj_w  = (const float*)d_in[11];
  const float* proj_b  = (const float*)d_in[12];
  const float* rel     = (const float*)d_in[13];
  const float* cv_g    = (const float*)d_in[14];
  const float* cv_b    = (const float*)d_in[15];
  const float* pwin_w  = (const float*)d_in[16];
  const float* pwin_b  = (const float*)d_in[17];
  const float* dw_w    = (const float*)d_in[18];
  const float* dw_b    = (const float*)d_in[19];
  const float* bn_g    = (const float*)d_in[20];
  const float* bn_b    = (const float*)d_in[21];
  const float* pwout_w = (const float*)d_in[22];
  const float* pwout_b = (const float*)d_in[23];
  const float* ff2_g   = (const float*)d_in[24];
  const float* ff2_b   = (const float*)d_in[25];
  const float* ff2_w1  = (const float*)d_in[26];
  const float* ff2_b1  = (const float*)d_in[27];
  const float* ff2_w2  = (const float*)d_in[28];
  const float* ff2_b2  = (const float*)d_in[29];
  const float* fin_g   = (const float*)d_in[30];
  const float* fin_b   = (const float*)d_in[31];
  float* xo = (float*)d_out;   // running residual buffer

  char* ws = (char*)d_ws;
  size_t off = 0;
  auto alloc = [&](size_t bytes) -> void* {
    size_t o = off;
    off += (bytes + 255) & ~(size_t)255;
    return (void*)(ws + o);
  };

  _Float16* wf_ff1w1 = (_Float16*)alloc((size_t)768 * 3072 * 2);
  _Float16* wf_ff1w2 = (_Float16*)alloc((size_t)3072 * 768 * 2);
  _Float16* wf_qkv   = (_Float16*)alloc((size_t)768 * 2304 * 2);
  _Float16* wf_proj  = (_Float16*)alloc((size_t)768 * 768 * 2);
  _Float16* wf_pwin  = (_Float16*)alloc((size_t)768 * 1536 * 2);  // transposed
  _Float16* wf_pwout = (_Float16*)alloc((size_t)768 * 768 * 2);   // transposed
  _Float16* wf_ff2w1 = (_Float16*)alloc((size_t)768 * 3072 * 2);
  _Float16* wf_ff2w2 = (_Float16*)alloc((size_t)3072 * 768 * 2);
  _Float16* ln16     = (_Float16*)alloc((size_t)ROWS * DIM * 2);
  void* bufA = alloc((size_t)ROWS * 3072 * 2);  // h16 (FFN) / mid32 (conv, same bytes)
  void* bufB = alloc((size_t)ROWS * 2304 * 2);  // qkv16 / convy f32
  void* bufC = alloc((size_t)ROWS * DIM * 2);   // attnout16 / y16
  float* g32    = (float*)alloc((size_t)ROWS * DIM * 4);
  float* bnMean = (float*)alloc(DIM * 4);
  float* bnInv  = (float*)alloc(DIM * 4);

  _Float16* h16       = (_Float16*)bufA;
  float*    mid32     = (float*)bufA;
  _Float16* qkv16     = (_Float16*)bufB;
  float*    convy     = (float*)bufB;
  _Float16* attnout16 = (_Float16*)bufC;
  _Float16* y16       = (_Float16*)bufC;

  // ---- weights -> f16 (transpose pwin/pwout so B is row-major (K,N))
  convert_w_kernel<<<2048, 256, 0, stream>>>(ff1_w1, wf_ff1w1, 768, 3072, 0);
  convert_w_kernel<<<2048, 256, 0, stream>>>(ff1_w2, wf_ff1w2, 3072, 768, 0);
  convert_w_kernel<<<2048, 256, 0, stream>>>(qkv_w,  wf_qkv,   768, 2304, 0);
  convert_w_kernel<<<2048, 256, 0, stream>>>(proj_w, wf_proj,  768, 768, 0);
  convert_w_kernel<<<2048, 256, 0, stream>>>(pwin_w, wf_pwin,  1536, 768, 1);
  convert_w_kernel<<<2048, 256, 0, stream>>>(pwout_w, wf_pwout, 768, 768, 1);
  convert_w_kernel<<<2048, 256, 0, stream>>>(ff2_w1, wf_ff2w1, 768, 3072, 0);
  convert_w_kernel<<<2048, 256, 0, stream>>>(ff2_w2, wf_ff2w2, 3072, 768, 0);

  // ---- FFN1 (half residual)
  layernorm_kernel<<<ROWS, 256, 0, stream>>>(x, ff1_g, ff1_b, ln16, nullptr);
  gemm_wmma_kernel<1><<<dim3(3072 / 64, ROWS / 128), 128, 0, stream>>>(
      ln16, wf_ff1w1, ff1_b1, nullptr, 0.f, h16, nullptr, 768, 3072);
  gemm_wmma_kernel<3><<<dim3(768 / 64, ROWS / 128), 128, 0, stream>>>(
      h16, wf_ff1w2, ff1_b2, x, 0.5f, nullptr, xo, 3072, 768);

  // ---- attention
  layernorm_kernel<<<ROWS, 256, 0, stream>>>(xo, at_g, at_b, ln16, nullptr);
  gemm_wmma_kernel<0><<<dim3(2304 / 64, ROWS / 128), 128, 0, stream>>>(
      ln16, wf_qkv, qkv_b, nullptr, 0.f, qkv16, nullptr, 768, 2304);
  attn_flash_kernel<<<dim3(SEQ / 128, BATCH * HEADS), 256, 0, stream>>>(
      qkv16, rel, attnout16);
  gemm_wmma_kernel<3><<<dim3(768 / 64, ROWS / 128), 128, 0, stream>>>(
      attnout16, wf_proj, proj_b, xo, 1.0f, nullptr, xo, 768, 768);

  // ---- conv module
  layernorm_kernel<<<ROWS, 256, 0, stream>>>(xo, cv_g, cv_b, ln16, nullptr);
  gemm_wmma_kernel<2><<<dim3(1536 / 64, ROWS / 128), 128, 0, stream>>>(
      ln16, wf_pwin, pwin_b, nullptr, 0.f, nullptr, mid32, 768, 1536);
  glu_kernel<<<2048, 256, 0, stream>>>(mid32, g32);
  dwconv_kernel<<<2048, 256, 0, stream>>>(g32, dw_w, dw_b, convy);
  bnstats_kernel<<<DIM, 256, 0, stream>>>(convy, bnMean, bnInv);
  bnsilu_kernel<<<2048, 256, 0, stream>>>(convy, bnMean, bnInv, bn_g, bn_b, y16);
  gemm_wmma_kernel<3><<<dim3(768 / 64, ROWS / 128), 128, 0, stream>>>(
      y16, wf_pwout, pwout_b, xo, 1.0f, nullptr, xo, 768, 768);

  // ---- FFN2 (half residual)
  layernorm_kernel<<<ROWS, 256, 0, stream>>>(xo, ff2_g, ff2_b, ln16, nullptr);
  gemm_wmma_kernel<1><<<dim3(3072 / 64, ROWS / 128), 128, 0, stream>>>(
      ln16, wf_ff2w1, ff2_b1, nullptr, 0.f, h16, nullptr, 768, 3072);
  gemm_wmma_kernel<3><<<dim3(768 / 64, ROWS / 128), 128, 0, stream>>>(
      h16, wf_ff2w2, ff2_b2, xo, 0.5f, nullptr, xo, 3072, 768);

  // ---- final LN (in place on d_out)
  layernorm_kernel<<<ROWS, 256, 0, stream>>>(xo, fin_g, fin_b, nullptr, xo);
}
